// SpatialTemporalGNN_9981503995971
// MI455X (gfx1250) — matrix operand order repaired and checked by reading, not compiled, not run on hardware
//
#include <hip/hip_runtime.h>
#include <hip/hip_bf16.h>

#define BB   128
#define TT   12
#define NN   325
#define FF   2
#define HH   64
#define NHH  4
#define DHH  16
#define DFFC 256
#define BN   (BB*NN)        // 41600  (2600 * 16)
#define BTN  (BB*TT*NN)     // 499200 (31200 * 16)
#define NP   336            // 21 * 16 (padded node dim, M tiles)
#define KP   352            // 11 * 32 (padded node dim, K tiles of graph conv)
#define WPB  4              // waves (16-row M tiles) per block

typedef __attribute__((ext_vector_type(16))) _Float16 v16h;
typedef __attribute__((ext_vector_type(8)))  _Float16 v8h;
typedef __attribute__((ext_vector_type(8)))  float    v8f;
typedef __attribute__((ext_vector_type(4)))  float    v4f;
typedef __attribute__((ext_vector_type(4)))  unsigned int tdm_u4;
typedef __attribute__((ext_vector_type(8)))  int          tdm_i8;
typedef __attribute__((ext_vector_type(4)))  int          tdm_i4;

// ---------------- WMMA fragment loaders (wave32 layouts, CDNA5 ISA 7.12.2) -----

// A (16x32 f16): M = lane&15 ; elements 0..7 -> K=k0+half*8+j ;
// elements 8..15 -> K=k0+16+half*8+(j-8). Two contiguous 8-element runs.
__device__ __forceinline__ v16h frag_a_f32(const float* A, int lda, int row, int k0, int half) {
  const float* p = A + (size_t)row * lda + k0 + half * 8;
  v4f x0 = *(const v4f*)(p);
  v4f x1 = *(const v4f*)(p + 4);
  v4f x2 = *(const v4f*)(p + 16);
  v4f x3 = *(const v4f*)(p + 20);
  v16h a;
#pragma unroll
  for (int i = 0; i < 4; ++i) {
    a[i]      = (_Float16)x0[i];
    a[4 + i]  = (_Float16)x1[i];
    a[8 + i]  = (_Float16)x2[i];
    a[12 + i] = (_Float16)x3[i];
  }
  return a;
}

__device__ __forceinline__ v16h frag_a_f16(const _Float16* A, int lda, int row, int k0, int half) {
  const _Float16* p = A + (size_t)row * lda + k0 + half * 8;
  v8h lo = *(const v8h*)(p);
  v8h hi = *(const v8h*)(p + 16);
  v16h a;
#pragma unroll
  for (int i = 0; i < 8; ++i) { a[i] = lo[i]; a[8 + i] = hi[i]; }
  return a;
}

// B (32x16 f16): N = lane&15 ; elements 0..7 -> K=k0+half*8+j ;
// elements 8..15 -> K=k0+16+half*8+(j-8). Source transposed [N][ld] f16.
__device__ __forceinline__ v16h frag_b_t16(const _Float16* BT, int ldb, int k0, int col, int half) {
  const _Float16* p = BT + (size_t)col * ldb + k0 + half * 8;
  v8h lo = *(const v8h*)(p);
  v8h hi = *(const v8h*)(p + 16);
  v16h b;
#pragma unroll
  for (int i = 0; i < 8; ++i) { b[i] = lo[i]; b[8 + i] = hi[i]; }
  return b;
}

// ---------------- GEMM 1: pre-transposed f16 weights staged via TDM ------------
// C[M x NT*16] = act(A[M x KT*32] @ B + bias), B given as f16 transposed [Nc][K].
// The weight panel is DMA'd into LDS by the Tensor Data Mover with pad_enable
// inserting 4 DWORDs (8 halves) after every K-row -> [Nc][K+8] swizzled layout,
// then consumed as two ds_load_b128 per fragment. 4 waves / block.
// CMODE: 0 = f32 row-major, 1 = f16 row-major, 2 = f16 transposed [N][ldc].
template <int NT, int KT, bool AH, int CMODE, bool RELU>
__global__ void __launch_bounds__(WPB * 32) gemm_tdm(
    const void* Aptr, int lda, long aBatch,
    const _Float16* WT,             // [NT*16][KT*32] f16, zero-padded columns
    void* Cptr, int ldc, long cBatch, int cMvalid, int cNvalid,
    const float* bias, int Mtiles) {
  constexpr int K    = KT * 32;
  constexpr int KPAD = K + 8;       // swizzle pad: 144B/528B stride spreads LDS banks
  constexpr int Nc   = NT * 16;
  __shared__ _Float16 sB[Nc * KPAD];

  const int tid = threadIdx.x;

#if __has_builtin(__builtin_amdgcn_tensor_load_to_lds)
  if (tid < 32) {  // wave 0 issues one TDM descriptor for the whole panel
    unsigned long long ga = (unsigned long long)(size_t)WT;
    unsigned ldsa = (unsigned)(size_t)(const void*)sB;  // flat addr low 32 = LDS offset
    constexpr unsigned totE  = (unsigned)Nc * K / 2;    // 4-byte elements
    constexpr unsigned padIv = (KT == 8) ? 6u : 4u;     // interval = K/2 DWORDs = 2^(padIv+1)
    tdm_u4 g0;
    g0[0] = 1u;                                         // count=1, user mode
    g0[1] = ldsa;                                       // lds_addr
    g0[2] = (unsigned)(ga & 0xFFFFFFFFu);               // global_addr[31:0]
    g0[3] = (unsigned)((ga >> 32) & 0x01FFFFFFu) | (2u << 30);  // addr[56:32] | type=2
    tdm_i8 g1;
    g1[0] = (int)((2u << 16) | (1u << 20) | (padIv << 22) | (3u << 25));
    //        data_size=4B     pad_enable    pad_interval     pad_amount=4 DWORDs
    g1[1] = (int)((totE & 0xFFFFu) << 16);              // tensor_dim0[15:0]
    g1[2] = (int)((totE >> 16) | (1u << 16));           // tensor_dim0[31:16] | tensor_dim1=1
    g1[3] = (int)(totE << 16);                          // tile_dim0 (fits 16 bits <= 8192)
    g1[4] = 0;                                          // tile_dim1=0, tile_dim2=0
    g1[5] = (int)totE;                                  // tensor_dim0_stride lo32
    g1[6] = 0;
    g1[7] = 0;
    tdm_i4 gz = {0, 0, 0, 0};
#if defined(__clang_major__) && __clang_major__ >= 23
    tdm_i8 g4 = {0, 0, 0, 0, 0, 0, 0, 0};
    __builtin_amdgcn_tensor_load_to_lds(g0, g1, gz, gz, g4, 0);
#else
    __builtin_amdgcn_tensor_load_to_lds(g0, g1, gz, gz, 0);
#endif
    __builtin_amdgcn_s_wait_tensorcnt(0);
  }
#else
  // fallback: cooperative 16B copies with pad insertion
  for (int ch = tid; ch < Nc * (K / 8); ch += WPB * 32) {
    int row = ch / (K / 8), w = ch % (K / 8);
    *(v8h*)&sB[row * KPAD + w * 8] = *(const v8h*)&WT[(size_t)row * K + w * 8];
  }
#endif
  __syncthreads();

  const int wave = tid >> 5;
  const int lane = tid & 31;
  const int lrow = lane & 15;
  const int half = lane >> 4;
  const int mt   = blockIdx.x * WPB + wave;
  if (mt >= Mtiles) return;
  const long bz = blockIdx.y;
  const int  m0 = mt * 16;

  const float*    Af32 = (const float*)Aptr + aBatch * bz;
  const _Float16* Af16 = (const _Float16*)Aptr + aBatch * bz;

  v8f acc[NT];
#pragma unroll
  for (int j = 0; j < NT; ++j)
#pragma unroll
    for (int i = 0; i < 8; ++i) acc[j][i] = 0.f;

#pragma unroll
  for (int kt = 0; kt < KT; ++kt) {
    const int k0 = kt * 32;
    if (kt + 1 < KT) {  // prefetch next A K-slab (global_prefetch_b8)
      if constexpr (AH)
        __builtin_prefetch(Af16 + (size_t)(m0 + lrow) * lda + k0 + 32, 0, 1);
      else
        __builtin_prefetch(Af32 + (size_t)(m0 + lrow) * lda + k0 + 32, 0, 1);
    }
    v16h a;
    if constexpr (AH) a = frag_a_f16(Af16, lda, m0 + lrow, k0, half);
    else              a = frag_a_f32(Af32, lda, m0 + lrow, k0, half);
#pragma unroll
    for (int j = 0; j < NT; ++j) {
      const int col = j * 16 + lrow;
      const _Float16* p = &sB[col * KPAD + k0 + half * 8];
      v8h lo = *(const v8h*)(p);
      v8h hi = *(const v8h*)(p + 16);
      v16h b;
#pragma unroll
      for (int i = 0; i < 8; ++i) { b[i] = lo[i]; b[8 + i] = hi[i]; }
      acc[j] = __builtin_amdgcn_wmma_f32_16x16x32_f16(false, a, false, b, (short)0, acc[j],
                                                      false, false);
    }
  }

  float*    Cf = (float*)Cptr + cBatch * bz;
  _Float16* Ch = (_Float16*)Cptr + cBatch * bz;
#pragma unroll
  for (int j = 0; j < NT; ++j) {
    const int col = j * 16 + lrow;
    float bv = 0.f;
    if (bias && col < cNvalid) bv = bias[col];
#pragma unroll
    for (int r = 0; r < 8; ++r) {
      const int row = m0 + r + half * 8;
      float v = acc[j][r] + bv;
      if constexpr (RELU) v = fmaxf(v, 0.f);
      if constexpr (CMODE == 2) {
        Ch[(size_t)col * ldc + row] = (_Float16)v;
      } else if constexpr (CMODE == 1) {
        if (row < cMvalid && col < cNvalid) Ch[(size_t)row * ldc + col] = (_Float16)v;
      } else {
        if (row < cMvalid && col < cNvalid) Cf[(size_t)row * ldc + col] = v;
      }
    }
  }
}

// ---------------- GEMM 2: f16 A (padded) x f16 transposed B --------------------
// Graph convolution hg[b] = A_norm(f16, padded) @ hwT[b]^T ; contiguous B loads.
template <int NT>
__global__ void __launch_bounds__(WPB * 32) gemm_t(
    const _Float16* Ah, int lda, long aBatch,
    const _Float16* Bt, int ldb, long bBatch,
    float* Cf, int ldc, long cBatch, int cMvalid, int Kiters, int Mtiles) {
  const int tid  = threadIdx.x;
  const int wave = tid >> 5;
  const int lane = tid & 31;
  const int lrow = lane & 15;
  const int half = lane >> 4;
  const int mt   = blockIdx.x * WPB + wave;
  if (mt >= Mtiles) return;
  const long bz = blockIdx.y;
  const int  m0 = mt * 16;

  const _Float16* A = Ah + aBatch * bz;
  const _Float16* B = Bt + bBatch * bz;

  v8f acc[NT];
#pragma unroll
  for (int j = 0; j < NT; ++j)
#pragma unroll
    for (int i = 0; i < 8; ++i) acc[j][i] = 0.f;

  for (int kt = 0; kt < Kiters; ++kt) {
    const int k0 = kt * 32;
    v16h a = frag_a_f16(A, lda, m0 + lrow, k0, half);
#pragma unroll
    for (int j = 0; j < NT; ++j) {
      v16h b = frag_b_t16(B, ldb, k0, j * 16 + lrow, half);
      acc[j] = __builtin_amdgcn_wmma_f32_16x16x32_f16(false, a, false, b, (short)0, acc[j],
                                                      false, false);
    }
  }

  float* C = Cf + cBatch * bz;
#pragma unroll
  for (int j = 0; j < NT; ++j) {
    const int col = j * 16 + lrow;
#pragma unroll
    for (int r = 0; r < 8; ++r) {
      const int row = m0 + r + half * 8;
      if (row < cMvalid) C[(size_t)row * ldc + col] = acc[j][r];
    }
  }
}

// ---------------- small VALU kernels -------------------------------------------

__global__ void k_fill0(float* p, long n) {
  long i = (long)blockIdx.x * blockDim.x + threadIdx.x;
  if (i < n) p[i] = 0.f;
}

// weight pre-transpose: W[slice][K][N] f32 -> WT[slice][Nc][K] f16 (cols >= N zeroed)
__global__ void k_wT(const float* W, _Float16* WT, int K, int N, int Nc) {
  int s = blockIdx.y;
  const float* Ws = W + (size_t)s * K * N;
  _Float16* WTs = WT + (size_t)s * Nc * K;
  for (int e = blockIdx.x * blockDim.x + threadIdx.x; e < K * Nc; e += gridDim.x * blockDim.x) {
    int k = e / Nc, c = e % Nc;  // consecutive tid -> consecutive c (coalesced read)
    WTs[(size_t)c * K + k] = (_Float16)((c < N) ? Ws[(size_t)k * N + c] : 0.f);
  }
}

__global__ void k_rowsum(const float* A, float* dinv) {
  int i = blockIdx.x * blockDim.x + threadIdx.x;
  if (i < NN) {
    float s = 1.f;  // +I diagonal
    for (int j = 0; j < NN; ++j) s += A[(size_t)i * NN + j];
    dinv[i] = rsqrtf(s);
  }
}

__global__ void k_anorm(const float* A, const float* dinv, float* An32, _Float16* A16) {
  int idx = blockIdx.x * blockDim.x + threadIdx.x;
  if (idx >= NP * KP) return;
  int i = idx / KP, j = idx % KP;
  float v = 0.f;
  if (i < NN && j < NN) {
    v = (A[(size_t)i * NN + j] + (i == j ? 1.f : 0.f)) * dinv[i] * dinv[j];
    An32[(size_t)i * NN + j] = v;
  }
  A16[idx] = (_Float16)v;
}

// xw[b,n,f] = x[b,t,n,:] @ W_s[t] + b_s[t]   (F=2)
__global__ void k_xw(const float* x, const float* Ws, const float* bs, float* xw, int t) {
  int idx = blockIdx.x * blockDim.x + threadIdx.x;
  if (idx >= BN * FF) return;
  int f = idx & 1, bn = idx >> 1;
  int b = bn / NN, n = bn % NN;
  const float* xp = x + (((size_t)b * TT + t) * NN + n) * FF;
  xw[idx] = xp[0] * Ws[f] + xp[1] * Ws[2 + f] + bs[f];
}

// xg[b,n,f] = sum_m A_norm[n,m] * xw[b,m,f]
__global__ void k_xconv(const float* An32, const float* xw, float* xg) {
  int idx = blockIdx.x * blockDim.x + threadIdx.x;
  if (idx >= BN * FF) return;
  int f = idx & 1, bn = idx >> 1;
  int b = bn / NN, n = bn % NN;
  const float* arow = An32 + (size_t)n * NN;
  const float* xwb  = xw + (size_t)b * NN * FF;
  float s = 0.f;
  for (int m = 0; m < NN; ++m) s += arow[m] * xwb[m * FF + f];
  xg[idx] = s;
}

// GRU update (gx inline, K=F=2) + sinusoidal PE fused into hidden-state store
__global__ void k_gru(const float* xg, const float* Wx, const float* bx, const float* ghb,
                      const float* hg, float* h, float* outb, int t) {
  int idx = blockIdx.x * blockDim.x + threadIdx.x;
  if (idx >= BN * HH) return;
  int j = idx & (HH - 1);
  int bn = idx >> 6;
  int b = bn / NN, n = bn % NN;
  float x0 = xg[bn * 2], x1 = xg[bn * 2 + 1];
  float rx = x0 * Wx[j]       + x1 * Wx[192 + j]       + bx[j];
  float zx = x0 * Wx[64 + j]  + x1 * Wx[192 + 64 + j]  + bx[64 + j];
  float nx = x0 * Wx[128 + j] + x1 * Wx[192 + 128 + j] + bx[128 + j];
  const float* g = ghb + (size_t)bn * 192;
  float hv = hg[(size_t)bn * HH + j];
  float r   = 1.f / (1.f + __expf(-(rx + g[j])));
  float z   = 1.f / (1.f + __expf(-(zx + g[64 + j])));
  float nn2 = tanhf(nx + r * g[128 + j]);
  float hn  = (1.f - z) * nn2 + z * hv;
  h[(size_t)bn * HH + j] = hn;
  float div = __expf(-(float)((j >> 1) << 1) * 0.14391156831f);  // ln(1e4)/64
  float ang = (float)t * div;
  float pe  = (j & 1) ? __cosf(ang) : __sinf(ang);
  outb[(((size_t)b * TT + t) * NN + n) * HH + j] = hn + pe;
}

// attention over time (T=12) per (b, n, head); q/k/v in f16
__global__ void k_attn(const _Float16* q, const _Float16* k, const _Float16* v, float* ctx) {
  int idx = blockIdx.x * blockDim.x + threadIdx.x;
  if (idx >= BB * NN * NHH) return;
  int h = idx & 3;
  int n = (idx >> 2) % NN;
  int b = idx / (NN * NHH);
  const size_t hoff = (size_t)h * DHH;
  for (int tq = 0; tq < TT; ++tq) {
    size_t qb = (((size_t)b * TT + tq) * NN + n) * HH + hoff;
    float qv[DHH];
#pragma unroll
    for (int d = 0; d < DHH; ++d) qv[d] = (float)q[qb + d];
    float sc[TT];
    float mx = -1e30f;
#pragma unroll
    for (int s = 0; s < TT; ++s) {
      size_t kb = (((size_t)b * TT + s) * NN + n) * HH + hoff;
      float dot = 0.f;
#pragma unroll
      for (int d = 0; d < DHH; ++d) dot += qv[d] * (float)k[kb + d];
      dot *= 0.25f;  // 1/sqrt(16)
      sc[s] = dot;
      mx = fmaxf(mx, dot);
    }
    float sum = 0.f;
#pragma unroll
    for (int s = 0; s < TT; ++s) { float e = __expf(sc[s] - mx); sc[s] = e; sum += e; }
    float inv = 1.f / sum;
    float o[DHH];
#pragma unroll
    for (int d = 0; d < DHH; ++d) o[d] = 0.f;
#pragma unroll
    for (int s = 0; s < TT; ++s) {
      size_t vb = (((size_t)b * TT + s) * NN + n) * HH + hoff;
      float w = sc[s];
#pragma unroll
      for (int d = 0; d < DHH; ++d) o[d] += w * (float)v[vb + d];
    }
#pragma unroll
    for (int d = 0; d < DHH; ++d) ctx[qb + d] = o[d] * inv;
  }
}

// io = LayerNorm(io + add) * g + b   (row length HH, in place)
__global__ void k_ln(float* io, const float* add, const float* g, const float* b, int rows) {
  int r = blockIdx.x * blockDim.x + threadIdx.x;
  if (r >= rows) return;
  float* prow = io + (size_t)r * HH;
  const float* arow = add + (size_t)r * HH;
  float xr[HH];
  float s = 0.f;
#pragma unroll
  for (int i = 0; i < HH; ++i) { xr[i] = prow[i] + arow[i]; s += xr[i]; }
  float mu = s * (1.f / HH);
  float vs = 0.f;
#pragma unroll
  for (int i = 0; i < HH; ++i) { float d = xr[i] - mu; vs += d * d; }
  float rs = rsqrtf(vs * (1.f / HH) + 1e-5f);
#pragma unroll
  for (int i = 0; i < HH; ++i) prow[i] = (xr[i] - mu) * rs * g[i] + b[i];
}

// ---------------- host orchestration -------------------------------------------

extern "C" void kernel_launch(void* const* d_in, const int* in_sizes, int n_in,
                              void* d_out, int out_size, void* d_ws, size_t ws_size,
                              hipStream_t stream) {
  (void)in_sizes; (void)n_in; (void)out_size; (void)ws_size;

  const float* x       = (const float*)d_in[0];
  const float* A       = (const float*)d_in[1];
  const float* W_s     = (const float*)d_in[2];
  const float* b_s     = (const float*)d_in[3];
  const float* W_hs    = (const float*)d_in[4];
  const float* b_hs    = (const float*)d_in[5];
  const float* gru_Wx  = (const float*)d_in[6];
  const float* gru_Wh  = (const float*)d_in[7];
  const float* gru_bx  = (const float*)d_in[8];
  const float* gru_bh  = (const float*)d_in[9];
  const float* Wq      = (const float*)d_in[10];
  const float* Wk      = (const float*)d_in[11];
  const float* Wv      = (const float*)d_in[12];
  const float* Wo      = (const float*)d_in[13];
  const float* ln1_g   = (const float*)d_in[14];
  const float* ln1_b   = (const float*)d_in[15];
  const float* ln2_g   = (const float*)d_in[16];
  const float* ln2_b   = (const float*)d_in[17];
  const float* ffn_W1  = (const float*)d_in[18];
  const float* ffn_b1  = (const float*)d_in[19];
  const float* ffn_W2  = (const float*)d_in[20];
  const float* ffn_b2  = (const float*)d_in[21];
  const float* head_W1 = (const float*)d_in[22];
  const float* head_b1 = (const float*)d_in[23];
  const float* head_W2 = (const float*)d_in[24];
  const float* head_b2 = (const float*)d_in[25];
  float* outp = (float*)d_out;

  // ---- workspace layout (256B aligned slabs) ----
  size_t off = 0;
  auto alloc = [&](size_t bytes) -> void* {
    void* p = (char*)d_ws + off;
    off += (bytes + 255) & ~(size_t)255;
    return p;
  };
  float*     dinv  = (float*)alloc((size_t)KP * 4);
  float*     An32  = (float*)alloc((size_t)NN * NN * 4);
  _Float16*  A16   = (_Float16*)alloc((size_t)NP * KP * 2);
  // pre-transposed f16 weights [Nc][K] (zero-padded cols), TDM staging sources
  _Float16*  WhsT    = (_Float16*)alloc((size_t)(TT - 1) * HH * HH * 2);
  _Float16*  gruWhT  = (_Float16*)alloc((size_t)TT * 192 * HH * 2);
  _Float16*  WqT     = (_Float16*)alloc((size_t)HH * HH * 2);
  _Float16*  WkT     = (_Float16*)alloc((size_t)HH * HH * 2);
  _Float16*  WvT     = (_Float16*)alloc((size_t)HH * HH * 2);
  _Float16*  WoT     = (_Float16*)alloc((size_t)HH * HH * 2);
  _Float16*  ffnW1T  = (_Float16*)alloc((size_t)DFFC * HH * 2);
  _Float16*  ffnW2T  = (_Float16*)alloc((size_t)HH * DFFC * 2);
  _Float16*  headW1T = (_Float16*)alloc((size_t)HH * HH * 2);
  _Float16*  headW2T = (_Float16*)alloc((size_t)16 * HH * 2);
  float*     xw    = (float*)alloc((size_t)BN * FF * 4);
  float*     xg    = (float*)alloc((size_t)BN * FF * 4);
  float*     h_cur = (float*)alloc(((size_t)BN + 16) * HH * 4);  // +16 rows: M-tile overread slack
  float*     hg    = (float*)alloc((size_t)BN * HH * 4);
  _Float16*  hwT   = (_Float16*)alloc((size_t)BB * HH * KP * 2); // (h@Whs+b)^T, K-padded
  float*     ghb   = (float*)alloc((size_t)BN * 192 * 4);
  float*     outb  = (float*)alloc((size_t)BTN * HH * 4);
  float*     ao    = (float*)alloc((size_t)BTN * HH * 4);
  _Float16*  q16   = (_Float16*)alloc((size_t)BTN * HH * 2);
  _Float16*  k16   = (_Float16*)alloc((size_t)BTN * HH * 2);
  _Float16*  v16   = (_Float16*)alloc((size_t)BTN * HH * 2);
  float*     ctx   = (float*)alloc((size_t)BTN * HH * 4);
  _Float16*  ffn16 = (_Float16*)q16;  // FFN hidden aliases dead q/k/v/ctx region

  const dim3 blk(WPB * 32, 1, 1);
  const int  MT_NODE = NP / 16;                       // 21
  const int  GB_NODE = (MT_NODE + WPB - 1) / WPB;     // 6
  const int  GB_BN   = (BN / 16) / WPB;               // 650
  const int  GB_BTN  = (BTN / 16) / WPB;              // 7800

  // ---- adjacency normalization ----
  k_rowsum<<<(NN + 63) / 64, 64, 0, stream>>>(A, dinv);
  k_anorm<<<(NP * KP + 255) / 256, 256, 0, stream>>>(A, dinv, An32, A16);

  // ---- weight pre-transpose / f16 conversion (once per launch) ----
  k_wT<<<dim3(16, TT - 1), 256, 0, stream>>>(W_hs, WhsT, HH, HH, HH);
  k_wT<<<dim3(48, TT), 256, 0, stream>>>(gru_Wh, gruWhT, HH, 192, 192);
  k_wT<<<dim3(16, 1), 256, 0, stream>>>(Wq, WqT, HH, HH, HH);
  k_wT<<<dim3(16, 1), 256, 0, stream>>>(Wk, WkT, HH, HH, HH);
  k_wT<<<dim3(16, 1), 256, 0, stream>>>(Wv, WvT, HH, HH, HH);
  k_wT<<<dim3(16, 1), 256, 0, stream>>>(Wo, WoT, HH, HH, HH);
  k_wT<<<dim3(64, 1), 256, 0, stream>>>(ffn_W1, ffnW1T, HH, DFFC, DFFC);
  k_wT<<<dim3(64, 1), 256, 0, stream>>>(ffn_W2, ffnW2T, DFFC, HH, HH);
  k_wT<<<dim3(16, 1), 256, 0, stream>>>(head_W1, headW1T, HH, HH, HH);
  k_wT<<<dim3(4, 1), 256, 0, stream>>>(head_W2, headW2T, HH, FF, 16);

  // ---- h0 = 0 ----
  k_fill0<<<(int)(((long)(BN + 16) * HH + 255) / 256), 256, 0, stream>>>(h_cur,
                                                                         (long)(BN + 16) * HH);

  // ---- temporal scan ----
  for (int t = 0; t < TT; ++t) {
    k_xw<<<(BN * FF + 255) / 256, 256, 0, stream>>>(x, W_s + (size_t)t * 4,
                                                    b_s + (size_t)t * 2, xw, t);
    k_xconv<<<(BN * FF + 255) / 256, 256, 0, stream>>>(An32, xw, xg);

    if (t == 0) {
      // W_hs_p[0] = 0 and h0 = 0  ->  hg = 0
      k_fill0<<<(int)(((long)BN * HH + 255) / 256), 256, 0, stream>>>(hg, (long)BN * HH);
    } else {
      // hwT[b] = (h[b] @ W_hs[t-1] + b_hs[t-1])^T  (f16 transposed, K-padded)
      gemm_tdm<4, 2, false, 2, false><<<dim3(GB_NODE, BB), blk, 0, stream>>>(
          h_cur, HH, (long)NN * HH,
          WhsT + (size_t)(t - 1) * HH * HH,
          hwT, KP, (long)HH * KP, NP, HH,
          b_hs + (size_t)(t - 1) * HH, MT_NODE);
      // hg[b] = A_norm @ hwT[b]^T
      gemm_t<4><<<dim3(GB_NODE, BB), blk, 0, stream>>>(
          A16, KP, 0L, hwT, KP, (long)HH * KP,
          hg, HH, (long)NN * HH, NN, KP / 32, MT_NODE);
    }

    // ghb = hg @ gru_Wh[t] + gru_bh[t]   (K=64, N=192)
    gemm_tdm<12, 2, false, 0, false><<<dim3(GB_BN, 1), blk, 0, stream>>>(
        hg, HH, 0L, gruWhT + (size_t)t * 192 * HH,
        ghb, 192, 0L, BN, 192, gru_bh + (size_t)t * 192, BN / 16);

    k_gru<<<(BN * HH + 255) / 256, 256, 0, stream>>>(
        xg, gru_Wx + (size_t)t * FF * 192, gru_bx + (size_t)t * 192, ghb, hg, h_cur, outb, t);
  }

  // ---- transformer over time ----
  gemm_tdm<4, 2, false, 1, false><<<dim3(GB_BTN, 1), blk, 0, stream>>>(
      outb, HH, 0L, WqT, q16, HH, 0L, BTN, HH, (const float*)0, BTN / 16);
  gemm_tdm<4, 2, false, 1, false><<<dim3(GB_BTN, 1), blk, 0, stream>>>(
      outb, HH, 0L, WkT, k16, HH, 0L, BTN, HH, (const float*)0, BTN / 16);
  gemm_tdm<4, 2, false, 1, false><<<dim3(GB_BTN, 1), blk, 0, stream>>>(
      outb, HH, 0L, WvT, v16, HH, 0L, BTN, HH, (const float*)0, BTN / 16);

  k_attn<<<(BB * NN * NHH + 255) / 256, 256, 0, stream>>>(q16, k16, v16, ctx);

  // ao = ctx @ Wo
  gemm_tdm<4, 2, false, 0, false><<<dim3(GB_BTN, 1), blk, 0, stream>>>(
      ctx, HH, 0L, WoT, ao, HH, 0L, BTN, HH, (const float*)0, BTN / 16);

  // outb = LN1(outb + ao)
  k_ln<<<(BTN + 127) / 128, 128, 0, stream>>>(outb, ao, ln1_g, ln1_b, BTN);

  // ffn16 = relu(outb @ W1 + b1)   (N=256, f16 out)
  gemm_tdm<16, 2, false, 1, true><<<dim3(GB_BTN, 1), blk, 0, stream>>>(
      outb, HH, 0L, ffnW1T, ffn16, DFFC, 0L, BTN, DFFC, ffn_b1, BTN / 16);

  // ao = ffn16 @ W2 + b2   (K=256, f16 A)
  gemm_tdm<4, 8, true, 0, false><<<dim3(GB_BTN, 1), blk, 0, stream>>>(
      ffn16, DFFC, 0L, ffnW2T, ao, HH, 0L, BTN, HH, ffn_b2, BTN / 16);

  // outb = LN2(outb + ao)
  k_ln<<<(BTN + 127) / 128, 128, 0, stream>>>(outb, ao, ln2_g, ln2_b, BTN);

  // ao = relu(outb @ head_W1 + head_b1)
  gemm_tdm<4, 2, false, 0, true><<<dim3(GB_BTN, 1), blk, 0, stream>>>(
      outb, HH, 0L, headW1T, ao, HH, 0L, BTN, HH, head_b1, BTN / 16);

  // d_out = ao @ head_W2 + head_b2   (N=2: zero-padded cols, masked store)
  gemm_tdm<1, 2, false, 0, false><<<dim3(GB_BTN, 1), blk, 0, stream>>>(
      ao, HH, 0L, headW2T, outp, FF, 0L, BTN, FF, head_b2, BTN / 16);
}
